// Encoder_35579509080069
// MI455X (gfx1250) — matrix operand order repaired and checked by reading, not compiled, not run on hardware
//
#include <hip/hip_runtime.h>
#include <hip/hip_bf16.h>
#include <math.h>

#define D_MODEL 1024
#define N_HEADS 16
#define HEAD_E  64
#define FFN_DIM 4096
#define BATCH   8
#define SEQ     1024
#define MROWS   (BATCH*SEQ)   // 8192

typedef unsigned short u16;
typedef unsigned int   u32;
typedef __attribute__((ext_vector_type(16))) __bf16 v16bf;
typedef __attribute__((ext_vector_type(8)))  float  v8f;
typedef __attribute__((ext_vector_type(4)))  u32    u32x4;

union Frag {
  u32   w[8];
  u32x4 q[2];
  v16bf b;
};

__device__ __forceinline__ u16 f2b(float f) {
  u32 u = __float_as_uint(f);
  u32 r = u + 0x7FFFu + ((u >> 16) & 1u);   // round-to-nearest-even
  return (u16)(r >> 16);
}

__device__ __forceinline__ v8f wmma_bf16(const Frag& a, const Frag& b, v8f c) {
  return __builtin_amdgcn_wmma_f32_16x16x32_bf16(false, a.b, false, b.b,
                                                 (short)0, c, false, false);
}

__device__ __forceinline__ float gelu_exact(float v) {
  return 0.5f * v * (1.0f + erff(v * 0.70710678118654752440f));
}

// ---- CDNA5 async global->LDS copy (per-lane 16B, tracked by ASYNCcnt) ----
__device__ __forceinline__ void async_b128(u32 lds_off, const void* gaddr) {
  asm volatile("global_load_async_to_lds_b128 %0, %1, off"
               :: "v"(lds_off), "v"(gaddr) : "memory");
}
template<int N>
__device__ __forceinline__ void wait_asynccnt() {
  asm volatile("s_wait_asynccnt %0" :: "i"(N) : "memory");
}

// ---- 16-lane xor reduction step via ds_swizzle (group-of-32: and|or|xor) ----
template<int MASK>
__device__ __forceinline__ float swz_xor(float v) {
  int r = __builtin_amdgcn_ds_swizzle(__float_as_int(v), (MASK << 10) | 0x1F);
  return __int_as_float(r);
}
__device__ __forceinline__ float redmax16(float v) {
  v = fmaxf(v, swz_xor<1>(v)); v = fmaxf(v, swz_xor<2>(v));
  v = fmaxf(v, swz_xor<4>(v)); v = fmaxf(v, swz_xor<8>(v));
  return v;
}
__device__ __forceinline__ float redsum16(float v) {
  v += swz_xor<1>(v); v += swz_xor<2>(v);
  v += swz_xor<4>(v); v += swz_xor<8>(v);
  return v;
}

// ---------------------------------------------------------------- converts
__global__ void k_f32_to_bf16(const float* __restrict__ in, u16* __restrict__ out, int n) {
  int i = blockIdx.x * blockDim.x + threadIdx.x;
  if (i < n) out[i] = f2b(in[i]);
}

// in: [K,N] f32 row-major  ->  out: [N,K] bf16 (transposed)
__global__ void k_transpose_bf16(const float* __restrict__ in, u16* __restrict__ out,
                                 int K, int N) {
  int i = blockIdx.x * blockDim.x + threadIdx.x;
  if (i >= K * N) return;
  int k = i / N, n = i - k * N;
  out[(size_t)n * K + k] = f2b(in[i]);
}

// ---------------------------------------------------------------- WMMA GEMM
// C[M,N] = A[M,K](bf16) * Wt[N,K](bf16, pre-transposed) + bias
// Weight tile (shared by all 8 waves) is staged via async->LDS, double-buffered.
// EPI: 0 = bias, 1 = bias+GELU(erf), 2 = bias+residual
// OUTM: 0 = f32, 1 = bf16, 2 = f32+bf16, 3 = bf16 stored [B,H,E,L] (for V)
template<int EPI, int OUTM>
__global__ __launch_bounds__(256) void k_gemm_wmma(
    const u16* __restrict__ A, const u16* __restrict__ Wt,
    const float* __restrict__ bias, const float* __restrict__ resid,
    float* __restrict__ outF, u16* __restrict__ outB,
    int M, int N, int K)
{
  __shared__ __align__(16) u16 ldsB[2][64][32];   // [buf][n-col][k], 8 KB

  const int wave = threadIdx.x >> 5;
  const int lane = threadIdx.x & 31;
  const int hb   = lane >> 4;
  const int l16  = lane & 15;
  const int n0 = blockIdx.x * 64;
  const int m0 = blockIdx.y * 128 + wave * 16;
  const int koffA = hb * 8;
  const int koffB = hb * 8;

  const u16* aRow = A + (size_t)(m0 + l16) * K;

  // async staging roles: wave w covers weight rows w*8..w*8+7, 4 chunks of 8 halfs
  const int brow = wave * 8 + (lane >> 2);
  const int bchk = (lane & 3) * 8;
  const u16* bBase = Wt + (size_t)(n0 + brow) * K + bchk;

  v8f z = {0.f,0.f,0.f,0.f,0.f,0.f,0.f,0.f};
  v8f acc[4] = {z, z, z, z};

  async_b128((u32)(size_t)&ldsB[0][brow][bchk], bBase);

  int ibuf = 0;
  for (int kb = 0; kb < K; kb += 32, ibuf ^= 1) {
    const bool more = (kb + 32) < K;
    if (more) async_b128((u32)(size_t)&ldsB[ibuf ^ 1][brow][bchk], bBase + kb + 32);

    Frag fa;
    fa.q[0] = *(const u32x4*)(aRow + kb + koffA);
    fa.q[1] = *(const u32x4*)(aRow + kb + koffA + 16);
    if (more) __builtin_prefetch(aRow + kb + 32, 0, 1);

    if (more) wait_asynccnt<1>(); else wait_asynccnt<0>();
    __syncthreads();

#pragma unroll
    for (int t = 0; t < 4; ++t) {
      Frag fb;
      fb.q[0] = *(const u32x4*)&ldsB[ibuf][t * 16 + l16][koffB];
      fb.q[1] = *(const u32x4*)&ldsB[ibuf][t * 16 + l16][koffB + 16];
      acc[t] = wmma_bf16(fa, fb, acc[t]);
    }
    __syncthreads();
  }

  // epilogue: C/D layout — lane<16: N=l16, M=r ; lane>=16: N=l16, M=8+r
#pragma unroll
  for (int t = 0; t < 4; ++t) {
    const int col = n0 + t * 16 + l16;
    const float bcol = bias[col];
#pragma unroll
    for (int r = 0; r < 8; ++r) {
      const int row = m0 + hb * 8 + r;
      const size_t idx = (size_t)row * N + col;
      float v = acc[t][r] + bcol;
      if (EPI == 1) v = gelu_exact(v);
      if (EPI == 2) v += resid[idx];
      if (OUTM == 0) outF[idx] = v;
      if (OUTM == 1) outB[idx] = f2b(v);
      if (OUTM == 2) { outF[idx] = v; outB[idx] = f2b(v); }
      if (OUTM == 3) {
        const int bI = row >> 10, lI = row & 1023;
        const int hI = col >> 6,  eI = col & 63;
        outB[(((size_t)bI * N_HEADS + hI) * HEAD_E + eI) * SEQ + lI] = f2b(v);
      }
    }
  }
}

// ---------------------------------------------------------------- flash attention
// q,k: bf16 [B, L, D] ; vT: bf16 [B, H, E, L] ; ctx out: bf16 [B, L, D]
// K/V tiles (shared by all 8 waves) staged via async->LDS, double-buffered.
__global__ __launch_bounds__(256) void k_attn_flash(
    const u16* __restrict__ qb, const u16* __restrict__ kb,
    const u16* __restrict__ vT, u16* __restrict__ ctxb)
{
  __shared__ __align__(16) u16 klds[2][32][64];   // [buf][key][e], 8 KB
  __shared__ __align__(16) u16 vlds[2][64][32];   // [buf][e][key], 8 KB
  __shared__ __align__(16) u16 plds[8][16][34];   // per-wave P tile, padded

  const int wave = threadIdx.x >> 5;
  const int lane = threadIdx.x & 31;
  const int hb   = lane >> 4;
  const int l16  = lane & 15;
  const int bh = blockIdx.y;
  const int b = bh >> 4, h = bh & 15;
  const int q0 = blockIdx.x * 128 + wave * 16;
  const int koffA = hb * 8, koffB = hb * 8;
  const float scale = 0.125f;        // 1/sqrt(64)

  // staging roles
  const int krow = wave * 4 + (lane >> 3);       // 0..31 (key)
  const int kchk = (lane & 7) * 8;               // halfs within 64
  const int erow = wave * 8 + (lane >> 2);       // 0..63 (e)
  const int vchk = (lane & 3) * 8;               // halfs within 32
  const u16* kBase = kb + ((size_t)(b * SEQ + krow)) * D_MODEL + h * HEAD_E + kchk;
  const u16* vBase = vT + ((size_t)(bh * HEAD_E + erow)) * SEQ + vchk;

  // Q fragments (16 x 64), resident for the whole loop
  const u16* qRow = qb + ((size_t)(b * SEQ + q0 + l16)) * D_MODEL + h * HEAD_E;
  Frag fq[2];
#pragma unroll
  for (int kk = 0; kk < 2; ++kk) {
    fq[kk].q[0] = *(const u32x4*)(qRow + kk * 32 + koffA);
    fq[kk].q[1] = *(const u32x4*)(qRow + kk * 32 + koffA + 16);
  }

  v8f z = {0.f,0.f,0.f,0.f,0.f,0.f,0.f,0.f};
  v8f o[4] = {z, z, z, z};
  float mrow[8], lrow[8];
#pragma unroll
  for (int r = 0; r < 8; ++r) { mrow[r] = -1e30f; lrow[r] = 0.f; }

  // prologue: stage j=0 tiles into buffer 0
  async_b128((u32)(size_t)&klds[0][krow][kchk], kBase);
  async_b128((u32)(size_t)&vlds[0][erow][vchk], vBase);

  int ibuf = 0;
  for (int j = 0; j < SEQ; j += 32, ibuf ^= 1) {
    const bool more = (j + 32) < SEQ;
    if (more) {
      async_b128((u32)(size_t)&klds[ibuf ^ 1][krow][kchk], kBase + (size_t)(j + 32) * D_MODEL);
      async_b128((u32)(size_t)&vlds[ibuf ^ 1][erow][vchk], vBase + j + 32);
    }
    if (more) wait_asynccnt<2>(); else wait_asynccnt<0>();
    __syncthreads();

    // ---- S = scale * Q @ K^T for 32 keys (two 16x16 tiles) from LDS
    v8f s[2] = {z, z};
#pragma unroll
    for (int st = 0; st < 2; ++st) {
#pragma unroll
      for (int kk = 0; kk < 2; ++kk) {
        Frag fk;
        fk.q[0] = *(const u32x4*)&klds[ibuf][st * 16 + l16][kk * 32 + koffB];
        fk.q[1] = *(const u32x4*)&klds[ibuf][st * 16 + l16][kk * 32 + koffB + 16];
        s[st] = wmma_bf16(fq[kk], fk, s[st]);
      }
    }

    // ---- online softmax (row reductions via ds_swizzle within 16-lane half)
    float alpha[8];
#pragma unroll
    for (int r = 0; r < 8; ++r) {
      float a0 = s[0][r] * scale, a1 = s[1][r] * scale;
      float v = redmax16(fmaxf(a0, a1));
      float nm = fmaxf(mrow[r], v);
      alpha[r] = __expf(mrow[r] - nm);
      mrow[r] = nm;
      float p0 = __expf(a0 - nm), p1 = __expf(a1 - nm);
      s[0][r] = p0; s[1][r] = p1;
      lrow[r] = lrow[r] * alpha[r] + redsum16(p0 + p1);
    }
#pragma unroll
    for (int t = 0; t < 4; ++t)
#pragma unroll
      for (int r = 0; r < 8; ++r) o[t][r] = o[t][r] * alpha[r];

    // ---- redistribute P (C layout -> A layout) via per-wave LDS tile
#pragma unroll
    for (int st = 0; st < 2; ++st)
#pragma unroll
      for (int r = 0; r < 8; ++r)
        plds[wave][hb * 8 + r][st * 16 + l16] = f2b(s[st][r]);
    __syncthreads();

    Frag fp;
#pragma unroll
    for (int jj = 0; jj < 8; ++jj) {
      const int kcol = koffA + 2 * jj + (jj >= 4 ? 8 : 0);
      fp.w[jj] = *(const u32*)&plds[wave][l16][kcol];
    }

    // ---- O += P @ V from LDS (V transposed: rows are E, contiguous over keys)
#pragma unroll
    for (int t = 0; t < 4; ++t) {
      Frag fv;
      fv.q[0] = *(const u32x4*)&vlds[ibuf][t * 16 + l16][koffB];
      fv.q[1] = *(const u32x4*)&vlds[ibuf][t * 16 + l16][koffB + 16];
      o[t] = wmma_bf16(fp, fv, o[t]);
    }
    __syncthreads();   // all reads of ibuf done before it is overwritten
  }

  // ---- normalize and write ctx (bf16, [B,L,D])
#pragma unroll
  for (int t = 0; t < 4; ++t)
#pragma unroll
    for (int r = 0; r < 8; ++r) {
      const int row = q0 + hb * 8 + r;
      const float val = o[t][r] / lrow[r];
      ctxb[((size_t)(b * SEQ + row)) * D_MODEL + h * HEAD_E + t * 16 + l16] = f2b(val);
    }
}

// ---------------------------------------------------------------- layernorm
__global__ __launch_bounds__(256) void k_layernorm(
    const float* __restrict__ x, const float* __restrict__ gamma,
    const float* __restrict__ beta, float* __restrict__ out)
{
  __shared__ float s1[256], s2[256];
  const int row = blockIdx.x, t = threadIdx.x;
  const float* p = x + (size_t)row * D_MODEL;
  float v[4];
#pragma unroll
  for (int i = 0; i < 4; ++i) v[i] = p[t + i * 256];
  float sum = v[0] + v[1] + v[2] + v[3];
  float sq  = v[0]*v[0] + v[1]*v[1] + v[2]*v[2] + v[3]*v[3];
  s1[t] = sum; s2[t] = sq;
  __syncthreads();
  for (int s = 128; s > 0; s >>= 1) {
    if (t < s) { s1[t] += s1[t + s]; s2[t] += s2[t + s]; }
    __syncthreads();
  }
  const float mu = s1[0] * (1.f / (float)D_MODEL);
  const float var = s2[0] * (1.f / (float)D_MODEL) - mu * mu;
  const float rs = rsqrtf(var + 1e-5f);
  float* op = out + (size_t)row * D_MODEL;
#pragma unroll
  for (int i = 0; i < 4; ++i) {
    const int c = t + i * 256;
    op[c] = (v[i] - mu) * rs * gamma[c] + beta[c];
  }
}

// ---------------------------------------------------------------- launch
extern "C" void kernel_launch(void* const* d_in, const int* in_sizes, int n_in,
                              void* d_out, int out_size, void* d_ws, size_t ws_size,
                              hipStream_t stream)
{
  const float* x  = (const float*)d_in[0];
  const float* Wq = (const float*)d_in[1];  const float* bq = (const float*)d_in[2];
  const float* Wk = (const float*)d_in[3];  const float* bk = (const float*)d_in[4];
  const float* Wv = (const float*)d_in[5];  const float* bv = (const float*)d_in[6];
  const float* Wo = (const float*)d_in[7];  const float* bo = (const float*)d_in[8];
  const float* W1 = (const float*)d_in[9];  const float* b1 = (const float*)d_in[10];
  const float* W2 = (const float*)d_in[11]; const float* b2 = (const float*)d_in[12];
  const float* gamma = (const float*)d_in[13];
  const float* beta  = (const float*)d_in[14];
  float* out = (float*)d_out;

  char* ws = (char*)d_ws;
  size_t off = 0;
  auto alloc = [&](size_t bytes) -> void* {
    void* p = ws + off;
    off += (bytes + 255) & ~(size_t)255;
    return p;
  };

  u16* xb   = (u16*)alloc((size_t)MROWS * D_MODEL * 2);
  u16* WqT  = (u16*)alloc((size_t)D_MODEL * D_MODEL * 2);
  u16* WkT  = (u16*)alloc((size_t)D_MODEL * D_MODEL * 2);
  u16* WvT  = (u16*)alloc((size_t)D_MODEL * D_MODEL * 2);
  u16* WoT  = (u16*)alloc((size_t)D_MODEL * D_MODEL * 2);
  u16* W1T  = (u16*)alloc((size_t)D_MODEL * FFN_DIM * 2);
  u16* W2T  = (u16*)alloc((size_t)FFN_DIM * D_MODEL * 2);
  u16* qbuf = (u16*)alloc((size_t)MROWS * D_MODEL * 2);
  u16* kbuf = (u16*)alloc((size_t)MROWS * D_MODEL * 2);
  u16* vTb  = (u16*)alloc((size_t)MROWS * D_MODEL * 2);
  u16* ctxb = (u16*)alloc((size_t)MROWS * D_MODEL * 2);
  float* x1 = (float*)alloc((size_t)MROWS * D_MODEL * 4);
  u16* x1b  = (u16*)alloc((size_t)MROWS * D_MODEL * 2);
  float* x2 = (float*)alloc((size_t)MROWS * D_MODEL * 4);
  u16* hbuf = qbuf;  // reuse 64MB q/k/v/ctx region for FFN hidden (deps done)

  const int TPB = 256;

  // convert x -> bf16
  {
    int n = MROWS * D_MODEL;
    k_f32_to_bf16<<<(n + TPB - 1) / TPB, TPB, 0, stream>>>(x, xb, n);
  }
  // transpose-convert weights to [N,K] bf16
  {
    int n = D_MODEL * D_MODEL;
    dim3 g((n + TPB - 1) / TPB);
    k_transpose_bf16<<<g, TPB, 0, stream>>>(Wq, WqT, D_MODEL, D_MODEL);
    k_transpose_bf16<<<g, TPB, 0, stream>>>(Wk, WkT, D_MODEL, D_MODEL);
    k_transpose_bf16<<<g, TPB, 0, stream>>>(Wv, WvT, D_MODEL, D_MODEL);
    k_transpose_bf16<<<g, TPB, 0, stream>>>(Wo, WoT, D_MODEL, D_MODEL);
    int n1 = D_MODEL * FFN_DIM;
    k_transpose_bf16<<<(n1 + TPB - 1) / TPB, TPB, 0, stream>>>(W1, W1T, D_MODEL, FFN_DIM);
    k_transpose_bf16<<<(n1 + TPB - 1) / TPB, TPB, 0, stream>>>(W2, W2T, FFN_DIM, D_MODEL);
  }

  // QKV projections
  {
    dim3 g(D_MODEL / 64, MROWS / 128);
    k_gemm_wmma<0, 1><<<g, TPB, 0, stream>>>(xb, WqT, bq, nullptr, nullptr, qbuf,
                                             MROWS, D_MODEL, D_MODEL);
    k_gemm_wmma<0, 1><<<g, TPB, 0, stream>>>(xb, WkT, bk, nullptr, nullptr, kbuf,
                                             MROWS, D_MODEL, D_MODEL);
    k_gemm_wmma<0, 3><<<g, TPB, 0, stream>>>(xb, WvT, bv, nullptr, nullptr, vTb,
                                             MROWS, D_MODEL, D_MODEL);
  }

  // flash attention
  {
    dim3 g(SEQ / 128, BATCH * N_HEADS);
    k_attn_flash<<<g, TPB, 0, stream>>>(qbuf, kbuf, vTb, ctxb);
  }

  // output projection + residual (x)
  {
    dim3 g(D_MODEL / 64, MROWS / 128);
    k_gemm_wmma<2, 2><<<g, TPB, 0, stream>>>(ctxb, WoT, bo, x, x1, x1b,
                                             MROWS, D_MODEL, D_MODEL);
  }
  // FFN1 + GELU
  {
    dim3 g(FFN_DIM / 64, MROWS / 128);
    k_gemm_wmma<1, 1><<<g, TPB, 0, stream>>>(x1b, W1T, b1, nullptr, nullptr, hbuf,
                                             MROWS, FFN_DIM, D_MODEL);
  }
  // FFN2 + residual (x1)
  {
    dim3 g(D_MODEL / 64, MROWS / 128);
    k_gemm_wmma<2, 0><<<g, TPB, 0, stream>>>(hbuf, W2T, b2, x1, x2, nullptr,
                                             MROWS, D_MODEL, FFN_DIM);
  }
  // LayerNorm
  k_layernorm<<<MROWS, TPB, 0, stream>>>(x2, gamma, beta, out);
}
